// Attention_52871047414344
// MI455X (gfx1250) — matrix operand order repaired and checked by reading, not compiled, not run on hardware
//
#include <hip/hip_runtime.h>
#include <hip/hip_bf16.h>

#define B_ 4
#define S_ 1024
#define D_ 2048
#define NH_ 32
#define NKV_ 8
#define HD_ 64
#define NREP_ 4

typedef __attribute__((ext_vector_type(16))) __bf16 v16bf;
typedef __attribute__((ext_vector_type(8)))  float  v8f;
typedef __attribute__((ext_vector_type(4)))  unsigned int u32x4;
typedef __attribute__((ext_vector_type(8)))  unsigned int u32x8;

union Frag { unsigned int u[8]; v16bf v; };

// ---------------------------------------------------------------------------
// CDNA5 async global->LDS copy (ASYNCcnt-tracked, bypasses VGPR staging).
// LDS offset = low 32 bits of the generic pointer (flat LDS aperture uses
// addr[31:0] as the LDS address, ISA 10.2).
// ---------------------------------------------------------------------------
__device__ __forceinline__ void async_ld_b128(void* lds_ptr, const void* gptr) {
    unsigned loff = (unsigned)(unsigned long long)lds_ptr;
    unsigned long long ga = (unsigned long long)gptr;
    asm volatile("global_load_async_to_lds_b128 %0, %1, off"
                 :: "v"(loff), "v"(ga) : "memory");
}
__device__ __forceinline__ void wait_async0() {
    asm volatile("s_wait_asynccnt 0" ::: "memory");
}

// ---------------------------------------------------------------------------
// CDNA5 Tensor Data Mover: 2D tile (rows of tile_d0 bf16, tile_d1 rows,
// row stride stride_elems) -> LDS, rows packed contiguously.
// D# group0/group1 per ISA 8.3/8.4: count=1, type=2, data_size=1 (2 bytes).
// Issued by one wave; tracked on TENSORcnt.
// ---------------------------------------------------------------------------
__device__ __forceinline__ void tdm_load_2d(void* lds_ptr, const void* gptr,
                                            unsigned tile_d0, unsigned tile_d1,
                                            unsigned stride_elems) {
    unsigned long long ga = (unsigned long long)gptr;
    unsigned lds_off = (unsigned)(unsigned long long)lds_ptr;
    u32x4 g0;
    g0[0] = 1u;                                           // count=1, user mode
    g0[1] = lds_off;                                      // lds_addr
    g0[2] = (unsigned)ga;                                 // global_addr[31:0]
    g0[3] = (unsigned)((ga >> 32) & 0x1FFFFFFull)         // global_addr[56:32]
          | (2u << 30);                                   // type=2 (image)
    u32x8 g1;
    unsigned td0 = stride_elems;   // tensor_dim0: full row length (no OOB)
    unsigned td1 = tile_d1;        // tensor_dim1: rows
    g1[0] = 0x10000u;                                     // data_size=1 (2B)
    g1[1] = (td0 & 0xFFFFu) << 16;                        // tensor_dim0[15:0]
    g1[2] = (td0 >> 16) | ((td1 & 0xFFFFu) << 16);        // d0 hi | d1 lo
    g1[3] = (td1 >> 16) | (tile_d0 << 16);                // d1 hi | tile_dim0
    g1[4] = tile_d1 & 0xFFFFu;                            // tile_dim1 (dim2=0)
    g1[5] = stride_elems;                                 // tensor_dim0_stride lo
    g1[6] = 0u;                                           // stride hi | dim1_stride lo
    g1[7] = 0u;
    asm volatile("tensor_load_to_lds %0, %1" :: "s"(g0), "s"(g1) : "memory");
}

// ---------------------------------------------------------------------------
// fp32 -> bf16 elementwise convert
// ---------------------------------------------------------------------------
__global__ __launch_bounds__(256) void cvt_f32_bf16(const float* __restrict__ src,
                                                    __bf16* __restrict__ dst,
                                                    unsigned long long n) {
    unsigned long long i = (unsigned long long)blockIdx.x * 256ull + threadIdx.x;
    if (i < n) dst[i] = (__bf16)src[i];
}

// ---------------------------------------------------------------------------
// WMMA GEMM:  C[M,N] = A[M,K] * Bt[N,K]^T   (A,Bt bf16 row-major)
// Block tile 64(M) x 256(N), BK=32, 8 waves, each wave 32x64 (2x4 WMMA tiles).
// Tiles staged to LDS by the Tensor Data Mover (one wave issues 2 D#s).
// ---------------------------------------------------------------------------
template <bool OUT_BF16>
__global__ __launch_bounds__(256) void gemm_xwt(const __bf16* __restrict__ A,
                                                const __bf16* __restrict__ Bt,
                                                void* __restrict__ Cout,
                                                int M, int N, int K) {
    __shared__ __bf16 As[64][32];    //  4 KB
    __shared__ __bf16 Bs[256][32];   // 16 KB
    const int tid  = threadIdx.x;
    const int wave = tid >> 5, lane = tid & 31;
    const int l16  = lane & 15, h2 = lane >> 4;
    const int bm = blockIdx.y * 64, bn = blockIdx.x * 256;
    const int wm = (wave & 1) * 32, wn = (wave >> 1) * 64;

    v8f acc[2][4] = {};

    for (int k0 = 0; k0 < K; k0 += 32) {
        if (wave == 0) {
            // A tile 64x32, B tile 256x32: two TDM descriptors, one wave
            tdm_load_2d(&As[0][0], &A[(size_t)bm * K + k0], 32, 64, (unsigned)K);
            tdm_load_2d(&Bs[0][0], &Bt[(size_t)bn * K + k0], 32, 256, (unsigned)K);
            __builtin_amdgcn_s_wait_tensorcnt(0);
        }
        __syncthreads();

        Frag af[2], bf[4];
        #pragma unroll
        for (int mt = 0; mt < 2; ++mt) {
            int m = wm + mt * 16 + l16;
            #pragma unroll
            for (int v = 0; v < 8; ++v) {
                int kk = (v < 4) ? (2 * v + 8 * h2) : (16 + 2 * (v - 4) + 8 * h2);
                af[mt].u[v] = *(const unsigned int*)&As[m][kk];
            }
        }
        #pragma unroll
        for (int nt = 0; nt < 4; ++nt) {
            int n = wn + nt * 16 + l16;
            #pragma unroll
            for (int v = 0; v < 8; ++v)
                bf[nt].u[v] = *(const unsigned int*)&Bs[n][(v >> 2) * 16 + 8 * h2 + 2 * (v & 3)];
        }
        #pragma unroll
        for (int mt = 0; mt < 2; ++mt)
            #pragma unroll
            for (int nt = 0; nt < 4; ++nt)
                acc[mt][nt] = __builtin_amdgcn_wmma_f32_16x16x32_bf16(
                    false, af[mt].v, false, bf[nt].v, (short)0, acc[mt][nt], false, false);
        __syncthreads();
    }

    #pragma unroll
    for (int mt = 0; mt < 2; ++mt)
        #pragma unroll
        for (int nt = 0; nt < 4; ++nt) {
            int n = bn + wn + nt * 16 + l16;
            #pragma unroll
            for (int v = 0; v < 8; ++v) {
                int m = bm + wm + mt * 16 + v + 8 * h2;
                float val = acc[mt][nt][v];
                if (OUT_BF16) ((__bf16*)Cout)[(size_t)m * N + n] = (__bf16)val;
                else          ((float*)Cout)[(size_t)m * N + n]  = val;
            }
        }
}

// ---------------------------------------------------------------------------
// RoPE on Q:  Qf (B*S, NH*HD) f32  ->  Qb (B,NH,S,HD) bf16
// ---------------------------------------------------------------------------
__global__ __launch_bounds__(256) void rope_q_kernel(const float* __restrict__ Qf,
                                                     const float* __restrict__ freqs,
                                                     __bf16* __restrict__ Qb) {
    unsigned long long idx = (unsigned long long)blockIdx.x * 256ull + threadIdx.x;
    int i = idx & 31; unsigned long long t = idx >> 5;
    int h = t % NH_; t /= NH_;
    int s = t % S_;  int b = t / S_;
    float f = freqs[s * 32 + i];
    float c = __cosf(f), sn = __sinf(f);
    const float* src = &Qf[((size_t)b * S_ + s) * D_ + h * HD_ + 2 * i];
    float t0 = src[0], t1 = src[1];
    float o0 = t0 * c - t1 * sn, o1 = t0 * sn + t1 * c;
    __bf16* dst = &Qb[(((size_t)b * NH_ + h) * S_ + s) * HD_ + 2 * i];
    dst[0] = (__bf16)o0; dst[1] = (__bf16)o1;
}

// ---------------------------------------------------------------------------
// RoPE on K: Kf (B*S, NKV*HD) f32 -> Kb (B,NKV,S,HD) bf16 + repeated f32 kout
// ---------------------------------------------------------------------------
__global__ __launch_bounds__(256) void rope_k_kernel(const float* __restrict__ Kf,
                                                     const float* __restrict__ freqs,
                                                     __bf16* __restrict__ Kb,
                                                     float* __restrict__ kout) {
    unsigned long long idx = (unsigned long long)blockIdx.x * 256ull + threadIdx.x;
    int i = idx & 31; unsigned long long t = idx >> 5;
    int kv = t % NKV_; t /= NKV_;
    int s = t % S_;    int b = t / S_;
    float f = freqs[s * 32 + i];
    float c = __cosf(f), sn = __sinf(f);
    const float* src = &Kf[((size_t)b * S_ + s) * (NKV_ * HD_) + kv * HD_ + 2 * i];
    float t0 = src[0], t1 = src[1];
    float o0 = t0 * c - t1 * sn, o1 = t0 * sn + t1 * c;
    __bf16* dst = &Kb[(((size_t)b * NKV_ + kv) * S_ + s) * HD_ + 2 * i];
    dst[0] = (__bf16)o0; dst[1] = (__bf16)o1;
    #pragma unroll
    for (int r = 0; r < NREP_; ++r) {
        float* d = &kout[(((size_t)b * NH_ + kv * NREP_ + r) * S_ + s) * HD_ + 2 * i];
        d[0] = o0; d[1] = o1;
    }
}

// ---------------------------------------------------------------------------
// V: Vf (B*S, NKV*HD) f32 -> Vb (B,NKV,S,HD) bf16 + repeated f32 vout
// ---------------------------------------------------------------------------
__global__ __launch_bounds__(256) void copy_v_kernel(const float* __restrict__ Vf,
                                                     __bf16* __restrict__ Vb,
                                                     float* __restrict__ vout) {
    unsigned long long idx = (unsigned long long)blockIdx.x * 256ull + threadIdx.x;
    int d = idx & (HD_ - 1); unsigned long long t = idx >> 6;
    int kv = t % NKV_; t /= NKV_;
    int s = t % S_;    int b = t / S_;
    float val = Vf[((size_t)b * S_ + s) * (NKV_ * HD_) + kv * HD_ + d];
    Vb[(((size_t)b * NKV_ + kv) * S_ + s) * HD_ + d] = (__bf16)val;
    #pragma unroll
    for (int r = 0; r < NREP_; ++r)
        vout[(((size_t)b * NH_ + kv * NREP_ + r) * S_ + s) * HD_ + d] = val;
}

// ---------------------------------------------------------------------------
// Flash attention: Qb(B,NH,S,HD), Kb/Vb(B,NKV,S,HD) bf16 -> Ob (B,S,NH*HD) bf16
// 8 waves/block, 16 queries/wave (128 q/block), 32-key LDS tiles, causal.
// K tile staged via async-to-LDS; V tile transposed in LDS for PV B-fragments.
// ---------------------------------------------------------------------------
__global__ __launch_bounds__(256) void flash_attn_kernel(const __bf16* __restrict__ Qb,
                                                         const __bf16* __restrict__ Kb,
                                                         const __bf16* __restrict__ Vb,
                                                         __bf16* __restrict__ Ob) {
    __shared__ __bf16 Kt[32][64];      // [key][d]
    __shared__ __bf16 Vt[64][32];      // [d][key]  (transposed for B-frag pairs)
    __shared__ __bf16 Ps[8][16][32];   // per-wave P scratch [q][key]

    const int tid  = threadIdx.x;
    const int wave = tid >> 5, lane = tid & 31;
    const int l16  = lane & 15, h2 = lane >> 4;

    int blk = blockIdx.x;
    int qt  = blk & 7;         // 8 query tiles of 128
    int bh  = blk >> 3;
    int h   = bh % NH_, b = bh / NH_;
    int kvh = h >> 2;          // NREP = 4
    int q0  = qt * 128;
    int qrow = q0 + wave * 16; // first query of this wave

    // Q fragments (16 x 64, two K-chunks of 32)
    const __bf16* Qp = Qb + ((size_t)b * NH_ + h) * S_ * HD_;
    Frag qf[2];
    #pragma unroll
    for (int c = 0; c < 2; ++c)
        #pragma unroll
        for (int v = 0; v < 8; ++v) {
            int kk = (v < 4) ? (2 * v + 8 * h2) : (16 + 2 * (v - 4) + 8 * h2);
            qf[c].u[v] = *(const unsigned int*)&Qp[(size_t)(qrow + l16) * HD_ + c * 32 + kk];
        }

    v8f o[4] = {};
    float mrow[8], lrow[8];
    #pragma unroll
    for (int v = 0; v < 8; ++v) { mrow[v] = -1e30f; lrow[v] = 0.0f; }

    const __bf16* Kp = Kb + ((size_t)b * NKV_ + kvh) * S_ * HD_;
    const __bf16* Vp = Vb + ((size_t)b * NKV_ + kvh) * S_ * HD_;

    const int ntiles = (q0 + 128) / 32;   // causal upper bound for this block
    for (int j = 0; j < ntiles; ++j) {
        int kt0 = j * 32;
        {
            int r = tid >> 3, c8 = (tid & 7) * 8;
            // K tile: async global->LDS, row-major
            async_ld_b128(&Kt[r][c8], &Kp[(size_t)(kt0 + r) * HD_ + c8]);
            // V tile: load + transpose into LDS
            const __bf16* vsrc = &Vp[(size_t)(kt0 + r) * HD_ + c8];
            #pragma unroll
            for (int e = 0; e < 8; ++e) Vt[c8 + e][r] = vsrc[e];
            wait_async0();
        }
        __syncthreads();

        // QK^T -> two 16x16 score tiles (keys kt0..+15, kt0+16..+31)
        v8f s[2];
        #pragma unroll
        for (int nt = 0; nt < 2; ++nt) {
            v8f z = {};
            #pragma unroll
            for (int c = 0; c < 2; ++c) {
                Frag bb;
                int key = nt * 16 + l16;
                #pragma unroll
                for (int v = 0; v < 8; ++v)
                    bb.u[v] = *(const unsigned int*)&Kt[key][c * 32 + (v >> 2) * 16 + 8 * h2 + 2 * (v & 3)];
                z = __builtin_amdgcn_wmma_f32_16x16x32_bf16(
                        false, qf[c].v, false, bb.v, (short)0, z, false, false);
            }
            s[nt] = z;
        }

        // scale + causal mask + online softmax (rows m = v + 8*h2, cols = l16)
        float mnew[8], alpha[8];
        #pragma unroll
        for (int v = 0; v < 8; ++v) {
            int q = qrow + v + 8 * h2;
            float s0 = s[0][v] * 0.125f;
            float s1 = s[1][v] * 0.125f;
            if (kt0 + l16 > q)      s0 = -1e30f;
            if (kt0 + 16 + l16 > q) s1 = -1e30f;
            s[0][v] = s0; s[1][v] = s1;
            float r = fmaxf(s0, s1);
            for (int off = 1; off < 16; off <<= 1)
                r = fmaxf(r, __shfl_xor(r, off, 32));
            mnew[v] = fmaxf(mrow[v], r);
        }
        #pragma unroll
        for (int v = 0; v < 8; ++v) {
            alpha[v] = __expf(mrow[v] - mnew[v]);
            float p0 = __expf(s[0][v] - mnew[v]);
            float p1 = __expf(s[1][v] - mnew[v]);
            s[0][v] = p0; s[1][v] = p1;
            float r = p0 + p1;
            for (int off = 1; off < 16; off <<= 1)
                r += __shfl_xor(r, off, 32);
            lrow[v] = lrow[v] * alpha[v] + r;
            mrow[v] = mnew[v];
        }
        #pragma unroll
        for (int t = 0; t < 4; ++t)
            #pragma unroll
            for (int v = 0; v < 8; ++v)
                o[t][v] *= alpha[v];

        // P (C layout) -> LDS -> reload as A fragment
        #pragma unroll
        for (int v = 0; v < 8; ++v) {
            Ps[wave][v + 8 * h2][l16]      = (__bf16)s[0][v];
            Ps[wave][v + 8 * h2][16 + l16] = (__bf16)s[1][v];
        }
        asm volatile("s_wait_dscnt 0" ::: "memory");  // wave-local LDS RAW

        Frag pf;
        #pragma unroll
        for (int v = 0; v < 8; ++v) {
            int kk = (v < 4) ? (2 * v + 8 * h2) : (16 + 2 * (v - 4) + 8 * h2);
            pf.u[v] = *(const unsigned int*)&Ps[wave][l16][kk];
        }
        // PV: O(16x64) += P(16x32) * V(32x64), 4 column tiles
        #pragma unroll
        for (int t = 0; t < 4; ++t) {
            Frag vb;
            int d = t * 16 + l16;
            #pragma unroll
            for (int v = 0; v < 8; ++v)
                vb.u[v] = *(const unsigned int*)&Vt[d][(v >> 2) * 16 + 8 * h2 + 2 * (v & 3)];
            o[t] = __builtin_amdgcn_wmma_f32_16x16x32_bf16(
                       false, pf.v, false, vb.v, (short)0, o[t], false, false);
        }
        __syncthreads();
    }

    // normalize, write (B, S, NH*HD) bf16 for the output projection
    #pragma unroll
    for (int t = 0; t < 4; ++t)
        #pragma unroll
        for (int v = 0; v < 8; ++v) {
            int q = qrow + v + 8 * h2;
            float val = o[t][v] / lrow[v];
            Ob[((size_t)b * S_ + q) * D_ + h * HD_ + t * 16 + l16] = (__bf16)val;
        }
}

// ---------------------------------------------------------------------------
extern "C" void kernel_launch(void* const* d_in, const int* in_sizes, int n_in,
                              void* d_out, int out_size, void* d_ws, size_t ws_size,
                              hipStream_t stream) {
    const float* x     = (const float*)d_in[0];
    const float* freqs = (const float*)d_in[1];
    // d_in[2] = mask (unused; causal computed analytically)
    const float* wq    = (const float*)d_in[3];
    const float* wk    = (const float*)d_in[4];
    const float* wv    = (const float*)d_in[5];
    const float* wo    = (const float*)d_in[6];

    float* out  = (float*)d_out;
    float* kout = out + (size_t)B_ * S_ * D_;
    float* vout = kout + (size_t)B_ * S_ * D_;

    const size_t MB = 1ull << 20;
    char* ws = (char*)d_ws;
    __bf16* xb    = (__bf16*)(ws + 0);          // 16 MB  (B*S, D)
    __bf16* wqb   = (__bf16*)(ws + 16 * MB);    //  8 MB  (D, D)
    __bf16* wkb   = (__bf16*)(ws + 24 * MB);    //  2 MB  (512, D)
    __bf16* wvb   = (__bf16*)(ws + 26 * MB);    //  2 MB
    __bf16* wob   = (__bf16*)(ws + 28 * MB);    //  8 MB
    float*  Qf    = (float*)(ws + 36 * MB);     // 32 MB  (B*S, D)
    float*  Kf    = (float*)(ws + 68 * MB);     //  8 MB  (B*S, 512)
    float*  Vf    = (float*)(ws + 76 * MB);     //  8 MB
    __bf16* Qb2   = (__bf16*)(ws + 84 * MB);    // 16 MB  (B,NH,S,HD)
    __bf16* Kb2   = (__bf16*)(ws + 100 * MB);   //  4 MB  (B,NKV,S,HD)
    __bf16* Vb2   = (__bf16*)(ws + 104 * MB);   //  4 MB
    __bf16* attnb = (__bf16*)(ws + 108 * MB);   // 16 MB  (B*S, D)

    auto cvt = [&](const float* s, __bf16* d, unsigned long long n) {
        cvt_f32_bf16<<<(unsigned)((n + 255) / 256), 256, 0, stream>>>(s, d, n);
    };
    cvt(x,  xb,  (unsigned long long)B_ * S_ * D_);
    cvt(wq, wqb, (unsigned long long)D_ * D_);
    cvt(wk, wkb, (unsigned long long)NKV_ * HD_ * D_);
    cvt(wv, wvb, (unsigned long long)NKV_ * HD_ * D_);
    cvt(wo, wob, (unsigned long long)D_ * D_);

    const int M = B_ * S_;  // 4096
    // Q = x @ wq^T : (4096, 2048)
    gemm_xwt<false><<<dim3(D_ / 256, M / 64), 256, 0, stream>>>(xb, wqb, Qf, M, D_, D_);
    // K, V : (4096, 512)
    gemm_xwt<false><<<dim3((NKV_ * HD_) / 256, M / 64), 256, 0, stream>>>(xb, wkb, Kf, M, NKV_ * HD_, D_);
    gemm_xwt<false><<<dim3((NKV_ * HD_) / 256, M / 64), 256, 0, stream>>>(xb, wvb, Vf, M, NKV_ * HD_, D_);

    // RoPE + layout + f32 k/v outputs
    rope_q_kernel<<<(B_ * S_ * NH_ * (HD_ / 2)) / 256, 256, 0, stream>>>(Qf, freqs, Qb2);
    rope_k_kernel<<<(B_ * S_ * NKV_ * (HD_ / 2)) / 256, 256, 0, stream>>>(Kf, freqs, Kb2, kout);
    copy_v_kernel<<<(B_ * S_ * NKV_ * HD_) / 256, 256, 0, stream>>>(Vf, Vb2, vout);

    // fused causal flash attention
    flash_attn_kernel<<<B_ * NH_ * (S_ / 128), 256, 0, stream>>>(Qb2, Kb2, Vb2, attnb);

    // out = attn @ wo^T : (4096, 2048) f32 straight into d_out
    gemm_xwt<false><<<dim3(D_ / 256, M / 64), 256, 0, stream>>>(attnb, wob, out, M, D_, D_);
}
